// LSTMClassifier_11948599017734
// MI455X (gfx1250) — matrix operand order repaired and checked by reading, not compiled, not run on hardware
//
#include <hip/hip_runtime.h>
#include <hip/hip_bf16.h>

// ---------------------------------------------------------------------------
// Problem constants:  B=64, T=1024, IN=128, H=512, OUT=128, OC=64, T'=1022.
// ---------------------------------------------------------------------------
#define Bsz   64
#define Tlen  1024
#define INC   128
#define Hdim  512
#define OUTD  128
#define OCC   64
#define TP    1022
#define NWG   32            // LSTM workgroups: each owns 512/32 = 16 hidden cols

typedef _Float16 v16h __attribute__((ext_vector_type(16)));
typedef _Float16 h8   __attribute__((ext_vector_type(8)));
typedef float    v8f  __attribute__((ext_vector_type(8)));
typedef unsigned int u32x4 __attribute__((ext_vector_type(4)));
typedef int          i32x8 __attribute__((ext_vector_type(8)));
typedef int          i32x4 __attribute__((ext_vector_type(4)));

// Workspace layout (bytes)
#define WS_CNT_OFF   0                                    // barrier counter
#define WS_SEQ_OFF   256                                  // f16 seq [TP][B][OC]
#define WS_SEQ_BYTES (TP * Bsz * OCC * 2)
#define WS_H16_OFF   (WS_SEQ_OFF + WS_SEQ_BYTES)          // f16 h [B][H]
#define WS_H16_BYTES (Bsz * Hdim * 2)
#define WS_H32_OFF   (WS_H16_OFF + WS_H16_BYTES)          // f32 h [B][H]

// LDS layout for LSTM kernel (dynamic; base of dynamic LDS is offset 0 —
// lstm_kernel declares no static __shared__)
#define WIH_PITCH 72            // 64 + 8 pad halves
#define WHH_PITCH 520           // 512 + 8 pad halves (1040 B = 65*16, aligned)
#define OFF_WIH   0
#define OFF_WHH   (OFF_WIH + 64 * WIH_PITCH * 2)          // 9216
#define OFF_GATES (OFF_WHH + 64 * WHH_PITCH * 2)          // 75776
#define OFF_BIAS  (OFF_GATES + 64 * 64 * 4)               // 92160
#define OFF_C     (OFF_BIAS + 64 * 4)                     // 92416
#define OFF_XT    (OFF_C + 64 * 16 * 4)                   // 96512  (8KB xt tile)
#define SMEM_BYTES (OFF_XT + Bsz * OCC * 2)               // 104704

#if defined(__has_builtin)
#if __has_builtin(__builtin_amdgcn_tensor_load_to_lds)
#define HAVE_TDM 1
#endif
#endif

// ---------------------------------------------------------------------------
// WMMA helpers (CDNA5 v_wmma_f32_16x16x32_f16, wave32)
// ---------------------------------------------------------------------------
__device__ __forceinline__ v8f wmma_f16(v16h a, v16h b, v8f c) {
  return __builtin_amdgcn_wmma_f32_16x16x32_f16(false, a, false, b,
                                                (short)0, c, false, false);
}

// A (M=16,K=32) fragment from row-major f16 [row][ld]:
// lane%16 = M row; per half-wave K-chunk base (lane/16)*8; VGPR0-3 hold
// K=kh..kh+7 (contiguous), VGPR4-7 hold K=16+kh..16+kh+7 (contiguous).
__device__ __forceinline__ v16h load_frag_A(const _Float16* p, int ld,
                                            int rowBase, int kBase, int lane) {
  const _Float16* r = p + (size_t)(rowBase + (lane & 15)) * ld + kBase +
                      ((lane >> 4) << 3);
  h8 lo = *(const h8*)(r);
  h8 hi = *(const h8*)(r + 16);
  return __builtin_shufflevector(lo, hi, 0, 1, 2, 3, 4, 5, 6, 7,
                                 8, 9, 10, 11, 12, 13, 14, 15);
}

// B (K=32,N=16) fragment from N-major f16 storage sW[n][k] (pitch ld):
// lane%16 = N; per half-wave K base (lane/16)*8; halves 0-7 hold
// K=kh..kh+7, halves 8-15 hold K=16+kh..16+kh+7 -> two 16-byte LDS loads.
__device__ __forceinline__ v16h load_frag_B(const _Float16* p, int ld,
                                            int nBase, int kBase, int lane) {
  const _Float16* r = p + (size_t)(nBase + (lane & 15)) * ld + kBase +
                      ((lane >> 4) << 3);
  h8 lo = *(const h8*)(r);
  h8 hi = *(const h8*)(r + 16);
  return __builtin_shufflevector(lo, hi, 0, 1, 2, 3, 4, 5, 6, 7,
                                 8, 9, 10, 11, 12, 13, 14, 15);
}

__device__ __forceinline__ float sigmoidf_(float x) {
  return 1.0f / (1.0f + __expf(-x));
}

// Monotone single-counter grid barrier (counter zeroed host-side per launch).
__device__ __forceinline__ void grid_barrier(unsigned* cnt, unsigned target) {
  __threadfence();
  __syncthreads();
  if (threadIdx.x == 0) {
    __hip_atomic_fetch_add(cnt, 1u, __ATOMIC_ACQ_REL, __HIP_MEMORY_SCOPE_AGENT);
    while (__hip_atomic_load(cnt, __ATOMIC_ACQUIRE, __HIP_MEMORY_SCOPE_AGENT) <
           target) {
      __builtin_amdgcn_s_sleep(1);
    }
  }
  __syncthreads();
}

// ---------------------------------------------------------------------------
// TDM: stage the contiguous 8KB xt tile (seq + t*8192) into LDS at OFF_XT.
// D# per CDNA5 ISA ch.8: group0 = {count, lds_addr, global_addr, type=2},
// group1 = {data_size=2B, tensor_dim0=4096, tensor_dim1=1, tile_dim0=4096,
// tile_dim1=1, strides=4096}.  Issued by wave 0 only (EXEC ignored by TDM).
// This toolchain exposes the 6-arg builtin:
//   (uint32x4 g0, int32x8 g1, int32x4 g2, int32x4 g3, int32x8, i32 cpol)
// ---------------------------------------------------------------------------
__device__ __forceinline__ void stage_xt(const _Float16* xt_global, int tid) {
#if defined(HAVE_TDM)
  if (tid < 32) {
    unsigned long long ga = (unsigned long long)(const void*)xt_global;
    u32x4 g0;
    g0[0] = 1u;                                   // count=1, user desc
    g0[1] = (unsigned)OFF_XT;                     // lds_addr
    g0[2] = (unsigned)(ga & 0xFFFFFFFFu);
    g0[3] = (unsigned)((ga >> 32) & 0x01FFFFFFu) | (2u << 30);  // type=2
    i32x8 g1;
    g1[0] = (int)(1u << 16);                      // data_size = 2 bytes
    g1[1] = (int)(4096u << 16);                   // tensor_dim0[15:0] = 4096
    g1[2] = (int)(1u << 16);                      // dim0 hi=0, tensor_dim1=1
    g1[3] = (int)(4096u << 16);                   // dim1 hi=0, tile_dim0=4096
    g1[4] = 1;                                    // tile_dim1=1, tile_dim2=0
    g1[5] = 4096;                                 // tensor_dim0_stride lo
    g1[6] = (int)(4096u << 16);                   // stride0 hi=0, stride1 lo=4096
    g1[7] = 0;
    i32x4 z4 = {0, 0, 0, 0};
    i32x8 z8 = {0, 0, 0, 0, 0, 0, 0, 0};
    __builtin_amdgcn_tensor_load_to_lds(g0, g1, z4, z4, z8, 0);
    __builtin_amdgcn_s_wait_tensorcnt(0);
  }
#else
  extern __shared__ char smem_[];
  unsigned* dst = (unsigned*)(smem_ + OFF_XT);
  const unsigned* src = (const unsigned*)xt_global;
  for (int i = tid; i < (Bsz * OCC * 2) / 4; i += 256) dst[i] = src[i];
#endif
  __syncthreads();
}

// ---------------------------------------------------------------------------
// Kernel 1: fused L2-normalize + conv1d(k=3, valid) + bias + ReLU.
// One 64-thread block per (b, t'); output seq f16 laid out [t'][b][oc].
// ---------------------------------------------------------------------------
__global__ void conv_norm_kernel(const float* __restrict__ inX,
                                 const float* __restrict__ conv_w,
                                 const float* __restrict__ conv_b,
                                 _Float16* __restrict__ seq) {
  const int idx = blockIdx.x;
  const int b = idx / TP;
  const int tp = idx % TP;
  const int tid = threadIdx.x;

  __shared__ float patch[3 * INC];
  __shared__ float red[64];

#pragma unroll
  for (int k = 0; k < 3; ++k) {
    const float* row = inX + ((size_t)b * Tlen + (tp + k)) * INC;
    float v0 = row[tid];
    float v1 = row[tid + 64];
    red[tid] = v0 * v0 + v1 * v1;
    __syncthreads();
#pragma unroll
    for (int s = 32; s > 0; s >>= 1) {
      if (tid < s) red[tid] += red[tid + s];
      __syncthreads();
    }
    const float scale = 1.0f / fmaxf(sqrtf(red[0]), 1e-12f);
    patch[k * INC + tid] = v0 * scale;
    patch[k * INC + tid + 64] = v1 * scale;
    __syncthreads();
  }

  const int oc = tid;
  const float* w = conv_w + (size_t)oc * INC * 3;   // [ic][k]
  float acc = conv_b[oc];
#pragma unroll 4
  for (int ic = 0; ic < INC; ++ic) {
    acc += patch[0 * INC + ic] * w[ic * 3 + 0];
    acc += patch[1 * INC + ic] * w[ic * 3 + 1];
    acc += patch[2 * INC + ic] * w[ic * 3 + 2];
  }
  acc = fmaxf(acc, 0.0f);
  seq[((size_t)tp * Bsz + b) * OCC + oc] = (_Float16)acc;
}

// ---------------------------------------------------------------------------
// Kernel 2: persistent WMMA LSTM. 32 WGs x 256 threads (8 wave32s).
// WG g owns hidden cols [g*16, g*16+16) => gate cols {q*512 + g*16 + j}.
// Weight slice lives in LDS for all 1022 steps; TDM stages xt; grid barrier
// per timestep.
// ---------------------------------------------------------------------------
__global__ void lstm_kernel(const float* __restrict__ w_ih,
                            const float* __restrict__ w_hh,
                            const float* __restrict__ b_ih,
                            const float* __restrict__ b_hh,
                            const _Float16* __restrict__ seq,
                            _Float16* __restrict__ h16,
                            float* __restrict__ h32,
                            unsigned* __restrict__ cnt) {
  extern __shared__ char smem[];
  _Float16* sWih = (_Float16*)(smem + OFF_WIH);     // [64][WIH_PITCH]
  _Float16* sWhh = (_Float16*)(smem + OFF_WHH);     // [64][WHH_PITCH]
  float* sGates  = (float*)(smem + OFF_GATES);      // [64 b][64 n]
  float* sBias   = (float*)(smem + OFF_BIAS);       // [64 n]
  float* sC      = (float*)(smem + OFF_C);          // [64 b][16 j]
  _Float16* sXt  = (_Float16*)(smem + OFF_XT);      // [64 b][64 oc]

  const int tid = threadIdx.x;
  const int lane = tid & 31;
  const int wave = tid >> 5;
  const int g = blockIdx.x;
  const int j0 = g * 16;

  // ---- load weight slices into LDS (f32 -> f16) ----
  for (int i = tid; i < 64 * 64; i += 256) {        // Wih slice [n][k=OC]
    const int n = i >> 6, k = i & 63;
    const int grow = ((n >> 4) << 9) + j0 + (n & 15);   // q*512 + j0 + j
    sWih[n * WIH_PITCH + k] = (_Float16)w_ih[(size_t)grow * OCC + k];
  }
  for (int i = tid; i < 64 * 512; i += 256) {       // Whh slice [n][k=H]
    const int n = i >> 9, k = i & 511;
    const int grow = ((n >> 4) << 9) + j0 + (n & 15);
    sWhh[n * WHH_PITCH + k] = (_Float16)w_hh[(size_t)grow * Hdim + k];
  }
  if (tid < 64) {
    const int grow = ((tid >> 4) << 9) + j0 + (tid & 15);
    sBias[tid] = b_ih[grow] + b_hh[grow];
  }
  for (int i = tid; i < 64 * 16; i += 256) sC[i] = 0.0f;
  for (int i = tid; i < 64 * 16; i += 256) {        // zero own h16 slice
    const int b = i >> 4, j = i & 15;
    h16[(size_t)b * Hdim + j0 + j] = (_Float16)0.0f;
  }
  __syncthreads();
  grid_barrier(cnt, (unsigned)NWG);                 // h zeroed device-wide

  // wave -> output tiles: M-tile m (batch rows m*16..), N-tiles nt0/nt1
  const int m = wave & 3;
  const int nt0 = (wave >> 2) * 2;
  const int nt1 = nt0 + 1;

  for (int t = 0; t < TP; ++t) {
    // stage xt tile to LDS via the Tensor Data Mover
    stage_xt(seq + (size_t)t * Bsz * OCC, tid);
    // prefetch next timestep's xt tile into cache while we compute
    if (t + 1 < TP && wave == 7) {
      const char* nxt = (const char*)(seq + (size_t)(t + 1) * Bsz * OCC);
      __builtin_prefetch(nxt + lane * 256, 0, 3);
    }

    v8f acc0 = {};
    v8f acc1 = {};

    // --- xt @ Wih^T : K = OC = 64 (A from LDS sXt) ---
#pragma unroll
    for (int kc = 0; kc < 2; ++kc) {
      v16h a = load_frag_A(sXt, OCC, m * 16, kc * 32, lane);
      v16h b0 = load_frag_B(sWih, WIH_PITCH, nt0 * 16, kc * 32, lane);
      v16h b1 = load_frag_B(sWih, WIH_PITCH, nt1 * 16, kc * 32, lane);
      acc0 = wmma_f16(a, b0, acc0);
      acc1 = wmma_f16(a, b1, acc1);
    }
    // --- h @ Whh^T : K = H = 512 ---
#pragma unroll
    for (int kc = 0; kc < 16; ++kc) {
      v16h a = load_frag_A(h16, Hdim, m * 16, kc * 32, lane);
      v16h b0 = load_frag_B(sWhh, WHH_PITCH, nt0 * 16, kc * 32, lane);
      v16h b1 = load_frag_B(sWhh, WHH_PITCH, nt1 * 16, kc * 32, lane);
      acc0 = wmma_f16(a, b0, acc0);
      acc1 = wmma_f16(a, b1, acc1);
    }

    // stage gate accumulators: C/D layout -> sGates[b][n]
    {
      const int row = m * 16 + ((lane >> 4) << 3);
      const int c0 = nt0 * 16 + (lane & 15);
      const int c1 = nt1 * 16 + (lane & 15);
#pragma unroll
      for (int v = 0; v < 8; ++v) {
        sGates[(row + v) * 64 + c0] = acc0[v];
        sGates[(row + v) * 64 + c1] = acc1[v];
      }
    }
    __syncthreads();

    // elementwise LSTM cell update for this WG's 64x16 slice
    for (int e = tid; e < Bsz * 16; e += 256) {
      const int b = e >> 4, j = e & 15;
      const float ig = sigmoidf_(sGates[b * 64 + j] + sBias[j]);
      const float fg = sigmoidf_(sGates[b * 64 + 16 + j] + sBias[16 + j]);
      const float gg = tanhf(sGates[b * 64 + 32 + j] + sBias[32 + j]);
      const float og = sigmoidf_(sGates[b * 64 + 48 + j] + sBias[48 + j]);
      const float c = fg * sC[e] + ig * gg;
      sC[e] = c;
      const float h = og * tanhf(c);
      h16[(size_t)b * Hdim + j0 + j] = (_Float16)h;
      if (t == TP - 1) h32[(size_t)b * Hdim + j0 + j] = h;
    }
    grid_barrier(cnt, (unsigned)(NWG * (t + 2)));
  }
}

// ---------------------------------------------------------------------------
// Kernel 3: out[b][o] = h32[b] . lin_w[o] + lin_b[o]   (64x128, K=512)
// ---------------------------------------------------------------------------
__global__ void head_kernel(const float* __restrict__ h32,
                            const float* __restrict__ lin_w,
                            const float* __restrict__ lin_b,
                            float* __restrict__ out) {
  const int e = blockIdx.x * blockDim.x + threadIdx.x;
  if (e >= Bsz * OUTD) return;
  const int b = e / OUTD, o = e % OUTD;
  const float* hr = h32 + (size_t)b * Hdim;
  const float* wr = lin_w + (size_t)o * Hdim;
  float acc = lin_b[o];
#pragma unroll 8
  for (int k = 0; k < Hdim; ++k) acc += hr[k] * wr[k];
  out[e] = acc;
}

// ---------------------------------------------------------------------------
extern "C" void kernel_launch(void* const* d_in, const int* in_sizes, int n_in,
                              void* d_out, int out_size, void* d_ws,
                              size_t ws_size, hipStream_t stream) {
  const float* inX    = (const float*)d_in[0];
  // d_in[1] = r (unused)
  const float* conv_w = (const float*)d_in[2];
  const float* conv_b = (const float*)d_in[3];
  const float* w_ih   = (const float*)d_in[4];
  const float* w_hh   = (const float*)d_in[5];
  const float* b_ih   = (const float*)d_in[6];
  const float* b_hh   = (const float*)d_in[7];
  const float* lin_w  = (const float*)d_in[8];
  const float* lin_b  = (const float*)d_in[9];

  char* ws = (char*)d_ws;
  unsigned* cnt  = (unsigned*)(ws + WS_CNT_OFF);
  _Float16* seq  = (_Float16*)(ws + WS_SEQ_OFF);
  _Float16* h16  = (_Float16*)(ws + WS_H16_OFF);
  float*    h32  = (float*)(ws + WS_H32_OFF);
  float*    out  = (float*)d_out;

  // reset grid-barrier counter (capturable async memset)
  (void)hipMemsetAsync(ws + WS_CNT_OFF, 0, 256, stream);

  conv_norm_kernel<<<Bsz * TP, 64, 0, stream>>>(inX, conv_w, conv_b, seq);

  static bool attr_set = [] {
    (void)hipFuncSetAttribute((const void*)lstm_kernel,
                              hipFuncAttributeMaxDynamicSharedMemorySize,
                              SMEM_BYTES);
    return true;
  }();
  (void)attr_set;

  lstm_kernel<<<NWG, 256, SMEM_BYTES, stream>>>(w_ih, w_hh, b_ih, b_hh, seq,
                                                h16, h32, cnt);

  head_kernel<<<(Bsz * OUTD + 255) / 256, 256, 0, stream>>>(h32, lin_w, lin_b,
                                                            out);
}
